// LSTM_Seq2Seq_35021163331980
// MI455X (gfx1250) — matrix operand order repaired and checked by reading, not compiled, not run on hardware
//
#include <hip/hip_runtime.h>

#define HIDDEN   512
#define LAGS     96
#define HORIZONS 24
#define NFEAT    32
#define BATCH    512

typedef __attribute__((ext_vector_type(16))) _Float16 v16h;
typedef __attribute__((ext_vector_type(8)))  _Float16 v8h;
typedef __attribute__((ext_vector_type(8)))  float    v8f;

__device__ __forceinline__ float sigmoid_fast(float x) {
    return 1.0f / (1.0f + __expf(-x));
}
__device__ __forceinline__ float tanh_fast(float x) {
    // 1 - 2/(e^{2x}+1): correct saturation at +/-inf, no NaN
    float e = __expf(2.0f * x);
    return 1.0f - 2.0f / (e + 1.0f);
}

// A fragment (16x32 f16, wave32 layout):
//  lane<16:  row=lane,     a[0..15]=K kb+0..15 (contiguous)
//  lane>=16: row=lane-16,  K shifted +16 (baked into 'alo' in the pointer)
__device__ __forceinline__ v16h load_a_frag(const _Float16* __restrict__ p, int kb) {
    v8h lo = *(const v8h*)(p + kb);
    v8h hi = *(const v8h*)(p + kb + 8);
    v16h a;
#pragma unroll
    for (int i = 0; i < 8; ++i) { a[i] = lo[i]; a[i + 8] = hi[i]; }
    return a;
}

#define WMMA_F16(A, B, C) \
    __builtin_amdgcn_wmma_f32_16x16x32_f16(false, (A), false, (B), (short)0, (C), false, false)

// ---------------------------------------------------------------------------
// Fused LSTM step: Z = x@Wih^T + h@Whh^T + bias ; gates ; c,h update.
// Wave tile: 32 batch rows (2 A-frags) x 16 units x 4 gates = 8 accumulators.
// The 4 gate B-fragments are reused across both M-tiles: 6 fragment loads (12 b128) / 8 WMMA.
// Block: 8 waves (4 along M, 2 along U) = 128 batch x 32 units.
// Grid: (512/128, 512/32) = (4, 16).
// ---------------------------------------------------------------------------
__global__ __launch_bounds__(256)
void lstm_step_kernel(const _Float16* __restrict__ x, int ldx, int Kin,
                      const _Float16* __restrict__ Wih,   // (2048, Kin) row-major f16
                      const _Float16* __restrict__ Whh,   // (2048, 512) row-major f16
                      const _Float16* __restrict__ h_in,  // (512, 512) f16
                      const float*    __restrict__ bias,  // (2048) = bih+bhh
                      float*          __restrict__ c,     // (512, 512) f32, in-place
                      _Float16*       __restrict__ h_out) // (512, 512) f16
{
    const int lane = threadIdx.x & 31;
    const int wave = threadIdx.x >> 5;
    const int m0 = blockIdx.x * 128 + (wave & 3) * 32;  // batch tile (2 x 16 rows)
    const int u0 = blockIdx.y * 32 + (wave >> 2) * 16;  // unit tile

    const int arow = lane & 15;
    const int alo  = (lane >> 4) << 4;   // A lane-group K offset (0 or 16)
    const int bn   = lane & 15;
    const int bko  = (lane >> 4) << 4;   // B lane-group K offset (0 or 16)

    v8f acc[2][4];
#pragma unroll
    for (int t = 0; t < 2; ++t)
#pragma unroll
        for (int g = 0; g < 4; ++g) acc[t][g] = (v8f){};

    // ---- x @ Wih^T ----
    {
        const _Float16* ap0 = x + (size_t)(m0 + arow) * ldx + alo;
        const _Float16* ap1 = x + (size_t)(m0 + 16 + arow) * ldx + alo;
        const _Float16* wi = Wih + (size_t)(u0 + 0 * HIDDEN + bn) * Kin + bko;
        const _Float16* wf = Wih + (size_t)(u0 + 1 * HIDDEN + bn) * Kin + bko;
        const _Float16* wg = Wih + (size_t)(u0 + 2 * HIDDEN + bn) * Kin + bko;
        const _Float16* wo = Wih + (size_t)(u0 + 3 * HIDDEN + bn) * Kin + bko;
        for (int kb = 0; kb < Kin; kb += 32) {
            v16h a0 = load_a_frag(ap0, kb);
            v16h a1 = load_a_frag(ap1, kb);
            v16h bi = *(const v16h*)(wi + kb);
            v16h bf = *(const v16h*)(wf + kb);
            v16h bg = *(const v16h*)(wg + kb);
            v16h bo = *(const v16h*)(wo + kb);
            acc[0][0] = WMMA_F16(a0, bi, acc[0][0]);
            acc[1][0] = WMMA_F16(a1, bi, acc[1][0]);
            acc[0][1] = WMMA_F16(a0, bf, acc[0][1]);
            acc[1][1] = WMMA_F16(a1, bf, acc[1][1]);
            acc[0][2] = WMMA_F16(a0, bg, acc[0][2]);
            acc[1][2] = WMMA_F16(a1, bg, acc[1][2]);
            acc[0][3] = WMMA_F16(a0, bo, acc[0][3]);
            acc[1][3] = WMMA_F16(a1, bo, acc[1][3]);
        }
    }
    // ---- h @ Whh^T (K = 512) ----
    {
        const _Float16* ap0 = h_in + (size_t)(m0 + arow) * HIDDEN + alo;
        const _Float16* ap1 = h_in + (size_t)(m0 + 16 + arow) * HIDDEN + alo;
        const _Float16* wi = Whh + (size_t)(u0 + 0 * HIDDEN + bn) * HIDDEN + bko;
        const _Float16* wf = Whh + (size_t)(u0 + 1 * HIDDEN + bn) * HIDDEN + bko;
        const _Float16* wg = Whh + (size_t)(u0 + 2 * HIDDEN + bn) * HIDDEN + bko;
        const _Float16* wo = Whh + (size_t)(u0 + 3 * HIDDEN + bn) * HIDDEN + bko;
        for (int kb = 0; kb < HIDDEN; kb += 32) {
            v16h a0 = load_a_frag(ap0, kb);
            v16h a1 = load_a_frag(ap1, kb);
            v16h bi = *(const v16h*)(wi + kb);
            v16h bf = *(const v16h*)(wf + kb);
            v16h bg = *(const v16h*)(wg + kb);
            v16h bo = *(const v16h*)(wo + kb);
            acc[0][0] = WMMA_F16(a0, bi, acc[0][0]);
            acc[1][0] = WMMA_F16(a1, bi, acc[1][0]);
            acc[0][1] = WMMA_F16(a0, bf, acc[0][1]);
            acc[1][1] = WMMA_F16(a1, bf, acc[1][1]);
            acc[0][2] = WMMA_F16(a0, bg, acc[0][2]);
            acc[1][2] = WMMA_F16(a1, bg, acc[1][2]);
            acc[0][3] = WMMA_F16(a0, bo, acc[0][3]);
            acc[1][3] = WMMA_F16(a1, bo, acc[1][3]);
        }
    }

    // ---- gates + cell update (C/D layout: VGPR v -> M = v + (lane>=16)*8, N = lane&15) ----
    const int u = u0 + (lane & 15);
    const float bi_u = bias[u];
    const float bf_u = bias[HIDDEN + u];
    const float bg_u = bias[2 * HIDDEN + u];
    const float bo_u = bias[3 * HIDDEN + u];
#pragma unroll
    for (int t = 0; t < 2; ++t) {
#pragma unroll
        for (int v = 0; v < 8; ++v) {
            int m = m0 + t * 16 + v + ((lane >> 4) << 3);
            float zi = acc[t][0][v] + bi_u;
            float zf = acc[t][1][v] + bf_u;
            float zg = acc[t][2][v] + bg_u;
            float zo = acc[t][3][v] + bo_u;
            size_t idx = (size_t)m * HIDDEN + u;
            float cn = sigmoid_fast(zf) * c[idx] + sigmoid_fast(zi) * tanh_fast(zg);
            c[idx] = cn;
            h_out[idx] = (_Float16)(sigmoid_fast(zo) * tanh_fast(cn));
        }
    }
}

// ---------------------------------------------------------------------------
// Generic small WMMA GEMM: out = X @ W^T + bias (f16 out). Used for fc4.
// ---------------------------------------------------------------------------
__global__ __launch_bounds__(256)
void gemm_f16_kernel(const _Float16* __restrict__ X, int ldx,
                     const _Float16* __restrict__ W, int ldw, int K,
                     const float* __restrict__ bias,
                     _Float16* __restrict__ out, int ldo)
{
    const int lane = threadIdx.x & 31;
    const int wave = threadIdx.x >> 5;
    const int m0 = blockIdx.x * 64 + (wave & 3) * 16;
    const int n0 = blockIdx.y * 32 + (wave >> 2) * 16;
    const int arow = lane & 15, alo = (lane >> 4) << 4;
    const int bn = lane & 15,  bko = (lane >> 4) << 4;

    v8f acc = {};
    const _Float16* ap = X + (size_t)(m0 + arow) * ldx + alo;
    const _Float16* bp = W + (size_t)(n0 + bn) * ldw + bko;
    for (int kb = 0; kb < K; kb += 32) {
        v16h a = load_a_frag(ap, kb);
        v16h b = *(const v16h*)(bp + kb);
        acc = WMMA_F16(a, b, acc);
    }
#pragma unroll
    for (int v = 0; v < 8; ++v) {
        int m = m0 + v + ((lane >> 4) << 3);
        int n = n0 + (lane & 15);
        out[(size_t)m * ldo + n] = (_Float16)(acc[v] + bias[n]);
    }
}

// fc1: out[b, t] = dot(h1[b,:], w) + b0 ;  one wave per batch row
__global__ void fc1_kernel(const _Float16* __restrict__ h, const float* __restrict__ w,
                           const float* __restrict__ b, float* __restrict__ out, int t)
{
    int row  = blockIdx.x;
    int lane = threadIdx.x;
    float s = 0.0f;
    for (int k = lane; k < HIDDEN; k += 32)
        s += (float)h[(size_t)row * HIDDEN + k] * w[k];
#pragma unroll
    for (int off = 16; off > 0; off >>= 1) s += __shfl_down(s, off, 32);
    if (lane == 0) out[(size_t)row * HORIZONS + t] = s + b[0];
}

__global__ void f32_to_f16_kernel(const float* __restrict__ in, _Float16* __restrict__ out, int n) {
    int i = blockIdx.x * blockDim.x + threadIdx.x;
    if (i < n) out[i] = (_Float16)in[i];
}
__global__ void bias_sum_kernel(const float* __restrict__ a, const float* __restrict__ b,
                                float* __restrict__ out, int n) {
    int i = blockIdx.x * blockDim.x + threadIdx.x;
    if (i < n) out[i] = a[i] + b[i];
}

// ---------------------------------------------------------------------------

static inline void cvt(const float* in, _Float16* out, int n, hipStream_t s) {
    f32_to_f16_kernel<<<(n + 255) / 256, 256, 0, s>>>(in, out, n);
}

extern "C" void kernel_launch(void* const* d_in, const int* in_sizes, int n_in,
                              void* d_out, int out_size, void* d_ws, size_t ws_size,
                              hipStream_t stream)
{
    // dict order: src, trg, train, enc..., dec..., fc1_w, fc1_b, fc4_w, fc4_b
    // (fallback to arg order if index 2 is not the scalar 'train')
    const int wb = (n_in > 2 && in_sizes[2] == 1) ? 3 : 2;
    const float* src      = (const float*)d_in[0];
    const float* enc_Wih0 = (const float*)d_in[wb + 0];
    const float* enc_Whh0 = (const float*)d_in[wb + 1];
    const float* enc_bih0 = (const float*)d_in[wb + 2];
    const float* enc_bhh0 = (const float*)d_in[wb + 3];
    const float* enc_Wih1 = (const float*)d_in[wb + 4];
    const float* enc_Whh1 = (const float*)d_in[wb + 5];
    const float* enc_bih1 = (const float*)d_in[wb + 6];
    const float* enc_bhh1 = (const float*)d_in[wb + 7];
    const float* dec_Wih0 = (const float*)d_in[wb + 8];
    const float* dec_Whh0 = (const float*)d_in[wb + 9];
    const float* dec_bih0 = (const float*)d_in[wb + 10];
    const float* dec_bhh0 = (const float*)d_in[wb + 11];
    const float* dec_Wih1 = (const float*)d_in[wb + 12];
    const float* dec_Whh1 = (const float*)d_in[wb + 13];
    const float* dec_bih1 = (const float*)d_in[wb + 14];
    const float* dec_bhh1 = (const float*)d_in[wb + 15];
    const float* fc1_w    = (const float*)d_in[wb + 16];
    const float* fc1_b    = (const float*)d_in[wb + 17];
    const float* fc4_w    = (const float*)d_in[wb + 18];
    const float* fc4_b    = (const float*)d_in[wb + 19];

    // ---- workspace bump allocator ----
    char* p = (char*)d_ws;
    auto alloc = [&](size_t bytes) -> void* {
        void* r = (void*)p;
        p += (bytes + 255) & ~(size_t)255;
        return r;
    };
    _Float16* src_h  = (_Float16*)alloc((size_t)BATCH * LAGS * NFEAT * 2);
    _Float16* eWih0  = (_Float16*)alloc((size_t)4 * HIDDEN * NFEAT * 2);
    _Float16* eWhh0  = (_Float16*)alloc((size_t)4 * HIDDEN * HIDDEN * 2);
    _Float16* eWih1  = (_Float16*)alloc((size_t)4 * HIDDEN * HIDDEN * 2);
    _Float16* eWhh1  = (_Float16*)alloc((size_t)4 * HIDDEN * HIDDEN * 2);
    _Float16* dWih0  = (_Float16*)alloc((size_t)4 * HIDDEN * NFEAT * 2);
    _Float16* dWhh0  = (_Float16*)alloc((size_t)4 * HIDDEN * HIDDEN * 2);
    _Float16* dWih1  = (_Float16*)alloc((size_t)4 * HIDDEN * HIDDEN * 2);
    _Float16* dWhh1  = (_Float16*)alloc((size_t)4 * HIDDEN * HIDDEN * 2);
    _Float16* fc4w_h = (_Float16*)alloc((size_t)NFEAT * HIDDEN * 2);
    float* bs_e0 = (float*)alloc(4 * HIDDEN * 4);
    float* bs_e1 = (float*)alloc(4 * HIDDEN * 4);
    float* bs_d0 = (float*)alloc(4 * HIDDEN * 4);
    float* bs_d1 = (float*)alloc(4 * HIDDEN * 4);
    _Float16* h0buf[2] = {(_Float16*)alloc((size_t)BATCH * HIDDEN * 2),
                          (_Float16*)alloc((size_t)BATCH * HIDDEN * 2)};
    _Float16* h1buf[2] = {(_Float16*)alloc((size_t)BATCH * HIDDEN * 2),
                          (_Float16*)alloc((size_t)BATCH * HIDDEN * 2)};
    float* c0 = (float*)alloc((size_t)BATCH * HIDDEN * 4);
    float* c1 = (float*)alloc((size_t)BATCH * HIDDEN * 4);
    _Float16* xdec = (_Float16*)alloc((size_t)BATCH * NFEAT * 2);

    // ---- precompute: f16 weights/inputs, fused biases, zeroed state ----
    cvt(src,      src_h,  BATCH * LAGS * NFEAT, stream);
    cvt(enc_Wih0, eWih0,  4 * HIDDEN * NFEAT,  stream);
    cvt(enc_Whh0, eWhh0,  4 * HIDDEN * HIDDEN, stream);
    cvt(enc_Wih1, eWih1,  4 * HIDDEN * HIDDEN, stream);
    cvt(enc_Whh1, eWhh1,  4 * HIDDEN * HIDDEN, stream);
    cvt(dec_Wih0, dWih0,  4 * HIDDEN * NFEAT,  stream);
    cvt(dec_Whh0, dWhh0,  4 * HIDDEN * HIDDEN, stream);
    cvt(dec_Wih1, dWih1,  4 * HIDDEN * HIDDEN, stream);
    cvt(dec_Whh1, dWhh1,  4 * HIDDEN * HIDDEN, stream);
    cvt(fc4_w,    fc4w_h, NFEAT * HIDDEN,      stream);
    bias_sum_kernel<<<8, 256, 0, stream>>>(enc_bih0, enc_bhh0, bs_e0, 4 * HIDDEN);
    bias_sum_kernel<<<8, 256, 0, stream>>>(enc_bih1, enc_bhh1, bs_e1, 4 * HIDDEN);
    bias_sum_kernel<<<8, 256, 0, stream>>>(dec_bih0, dec_bhh0, bs_d0, 4 * HIDDEN);
    bias_sum_kernel<<<8, 256, 0, stream>>>(dec_bih1, dec_bhh1, bs_d1, 4 * HIDDEN);
    hipMemsetAsync(h0buf[0], 0, (size_t)BATCH * HIDDEN * 2, stream);
    hipMemsetAsync(h1buf[0], 0, (size_t)BATCH * HIDDEN * 2, stream);
    hipMemsetAsync(c0,       0, (size_t)BATCH * HIDDEN * 4, stream);
    hipMemsetAsync(c1,       0, (size_t)BATCH * HIDDEN * 4, stream);

    const dim3 blk(256);
    const dim3 grd(BATCH / 128, HIDDEN / 32);  // (4, 16)
    const dim3 grd4(BATCH / 64, NFEAT / 32);   // (8, 1)
    int p0 = 0, p1 = 0;

    // ---- encoder: 96 steps x 2 layers ----
    for (int t = 0; t < LAGS; ++t) {
        lstm_step_kernel<<<grd, blk, 0, stream>>>(
            src_h + (size_t)t * NFEAT, LAGS * NFEAT, NFEAT,
            eWih0, eWhh0, h0buf[p0], bs_e0, c0, h0buf[p0 ^ 1]);
        p0 ^= 1;
        lstm_step_kernel<<<grd, blk, 0, stream>>>(
            h0buf[p0], HIDDEN, HIDDEN,
            eWih1, eWhh1, h1buf[p1], bs_e1, c1, h1buf[p1 ^ 1]);
        p1 ^= 1;
    }

    // ---- decoder: 24 autoregressive steps ----
    for (int t = 0; t < HORIZONS; ++t) {
        const _Float16* xin = (t == 0) ? (src_h + (size_t)(LAGS - 1) * NFEAT) : xdec;
        const int ldx = (t == 0) ? (LAGS * NFEAT) : NFEAT;
        lstm_step_kernel<<<grd, blk, 0, stream>>>(
            xin, ldx, NFEAT, dWih0, dWhh0, h0buf[p0], bs_d0, c0, h0buf[p0 ^ 1]);
        p0 ^= 1;
        lstm_step_kernel<<<grd, blk, 0, stream>>>(
            h0buf[p0], HIDDEN, HIDDEN, dWih1, dWhh1, h1buf[p1], bs_d1, c1, h1buf[p1 ^ 1]);
        p1 ^= 1;
        fc1_kernel<<<BATCH, 32, 0, stream>>>(h1buf[p1], fc1_w, fc1_b, (float*)d_out, t);
        gemm_f16_kernel<<<grd4, blk, 0, stream>>>(h1buf[p1], HIDDEN, fc4w_h, HIDDEN, HIDDEN,
                                                  fc4_b, xdec, NFEAT);
    }
    (void)ws_size; (void)out_size; (void)n_in;
}
